// QuantizedLinearINT4_30855045054753
// MI455X (gfx1250) — matrix operand order repaired and checked by reading, not compiled, not run on hardware
//
#include <hip/hip_runtime.h>

// ---- problem constants (match reference) ----
#define K_DIM   4096
#define N_DIM   11008
#define GROUPSZ 128

// ---- tiling ----
#define BM 256
#define BN 128
#define BK 64
#define LDSB 72   // 64 + 8 halves pad -> 144B row stride (16B aligned, bank-rotating)

typedef _Float16 half2v __attribute__((ext_vector_type(2)));
typedef _Float16 half8  __attribute__((ext_vector_type(8)));
typedef _Float16 half16 __attribute__((ext_vector_type(16)));
typedef float    float8 __attribute__((ext_vector_type(8)));

// addrspace(3) view of a shared-memory element, for extracting the LDS byte offset
typedef const _Float16 __attribute__((address_space(3))) lds_ch;

__global__ __launch_bounds__(256)
void q4gemm_wmma_f16(const _Float16* __restrict__ x,
                     const int*      __restrict__ wq,      // [N, K/2] one packed byte per int32
                     const _Float16* __restrict__ scales,  // [N, K/GROUPSZ]
                     const _Float16* __restrict__ bias,    // [N]
                     float*          __restrict__ out)     // [M, N] fp32
{
    __shared__ _Float16 As[2][BM][LDSB];
    __shared__ _Float16 Bs[2][BN][LDSB];

    const int tid   = threadIdx.x;
    const int lane  = tid & 31;
    const int wid   = tid >> 5;
    const int waveM = wid & 3;        // 4 waves along M (4*64 = 256)
    const int waveN = wid >> 2;       // 2 waves along N (2*64 = 128)

    const int n0 = blockIdx.x * BN;
    const int m0 = blockIdx.y * BM;

    // staging mapping: 8 chunks of 8 halves across BK, 32 rows/pass
    const int chunk = tid & 7;        // 0..7
    const int rowb  = tid >> 3;       // 0..31

    const int NT = K_DIM / BK;        // 64 k-tiles

    int4      wReg[4];
    _Float16  sReg[4];

    // A tile (256 rows): async DMA global -> LDS (no VGPR round trip, ASYNCcnt-tracked)
    auto asyncA = [&](int kt, int buf) {
        const int k0 = kt * BK;
        #pragma unroll
        for (int r = 0; r < 8; ++r) {
            const int row = rowb + r * 32;
            const uint32_t voff =
                (uint32_t)(((size_t)(m0 + row) * K_DIM + k0 + chunk * 8) * sizeof(_Float16));
            const uint32_t ldst =
                (uint32_t)(size_t)(lds_ch*)&As[buf][row][chunk * 8];
            asm volatile("global_load_async_to_lds_b128 %0, %1, %2"
                         :: "v"(ldst), "v"(voff), "s"(x) : "memory");
        }
    };

    auto loadW = [&](int kt) {
        const int k0 = kt * BK;
        #pragma unroll
        for (int r = 0; r < 4; ++r) {
            const int row = rowb + r * 32;
            wReg[r] = *(const int4*)(wq + (size_t)(n0 + row) * (K_DIM/2) + (k0 >> 1) + chunk * 4);
            sReg[r] = scales[(size_t)(n0 + row) * (K_DIM / GROUPSZ) + (k0 >> 7)];
        }
    };

    auto storeW = [&](int buf) {
        #pragma unroll
        for (int r = 0; r < 4; ++r) {
            const int row = rowb + r * 32;
            // int4 -> fp16 via exponent-bias trick:
            //   bitcast<f16>(0x6400 | nibble) == 1024 + nibble (exact)
            //   d = h*s + (-1032*s) == (nibble - 8) * s   (one v_pk_fma_f16 per pair)
            const _Float16 s = sReg[r];
            const half2v s2 = { s, s };
            const _Float16 cs = (_Float16)(-1032.0f) * s;
            const half2v c2 = { cs, cs };

            const int* pw = (const int*)&wReg[r];
            uint32_t u[4];
            #pragma unroll
            for (int j = 0; j < 4; ++j) {
                const uint32_t w  = (uint32_t)pw[j];
                const uint32_t pk = 0x64006400u | (w & 0x0Fu) | ((w & 0xF0u) << 12);
                half2v h = __builtin_bit_cast(half2v, pk);
                half2v d = h * s2 + c2;                 // -> v_pk_fma_f16
                u[j] = __builtin_bit_cast(uint32_t, d);
            }
            uint4 uv = { u[0], u[1], u[2], u[3] };
            *(half8*)&Bs[buf][row][chunk * 8] = __builtin_bit_cast(half8, uv);
        }
    };

    float8 acc[4][4] = {};

    // pipeline prologue
    asyncA(0, 0);
    loadW(0);
    storeW(0);
    loadW(1);
    asm volatile("s_wait_asynccnt 0x0" ::: "memory");
    __syncthreads();

    for (int kt = 0; kt < NT; ++kt) {
        const int cur = kt & 1;
        if (kt + 1 < NT) {
            asyncA(kt + 1, cur ^ 1);            // DMA next A tile into other buffer
            storeW(cur ^ 1);                    // dequant next W tile from regs
            if (kt + 2 < NT) loadW(kt + 2);     // issue next W global loads
        }

        #pragma unroll
        for (int ks = 0; ks < 2; ++ks) {        // two K=32 WMMA steps per BK=64
            const int kbA = ks * 32 + ((lane >> 4) * 8);   // A: hi lane-half holds k+8 / k+24
            const int kbB = ks * 32 + ((lane >> 4) * 8);   // B: hi lane-half holds k+8 / k+24

            half16 afrag[4];
            #pragma unroll
            for (int t = 0; t < 4; ++t) {
                const int row = waveM * 64 + t * 16 + (lane & 15);
                half8 lo = *(const half8*)&As[cur][row][kbA];
                half8 hi = *(const half8*)&As[cur][row][kbA + 16];
                afrag[t] = __builtin_shufflevector(lo, hi,
                           0,1,2,3,4,5,6,7,8,9,10,11,12,13,14,15);
            }
            #pragma unroll
            for (int u = 0; u < 4; ++u) {
                const int row = waveN * 64 + u * 16 + (lane & 15);
                half8 lo = *(const half8*)&Bs[cur][row][kbB];
                half8 hi = *(const half8*)&Bs[cur][row][kbB + 16];
                half16 bfrag = __builtin_shufflevector(lo, hi,
                               0,1,2,3,4,5,6,7,8,9,10,11,12,13,14,15);
                #pragma unroll
                for (int t = 0; t < 4; ++t) {
                    acc[t][u] = __builtin_amdgcn_wmma_f32_16x16x32_f16(
                        false, afrag[t], false, bfrag,
                        (short)0, acc[t][u], false, false);
                }
            }
        }

        if (kt + 1 < NT)
            asm volatile("s_wait_asynccnt 0x0" ::: "memory");  // next A tile landed in LDS
        __syncthreads();
    }

    // epilogue: C/D layout -> lanes 0-15: M=r, lanes 16-31: M=8+r; N = lane&15
    // Output is streamed once: non-temporal stores keep x/W resident in L2.
    #pragma unroll
    for (int u = 0; u < 4; ++u) {
        const int n = n0 + waveN * 64 + u * 16 + (lane & 15);
        const float bv = (float)bias[n];
        #pragma unroll
        for (int t = 0; t < 4; ++t) {
            #pragma unroll
            for (int r = 0; r < 8; ++r) {
                const int m = m0 + waveM * 64 + t * 16 + ((lane >> 4) * 8) + r;
                __builtin_nontemporal_store(acc[t][u][r] + bv,
                                            out + (size_t)m * N_DIM + n);
            }
        }
    }
}

extern "C" void kernel_launch(void* const* d_in, const int* in_sizes, int n_in,
                              void* d_out, int out_size, void* d_ws, size_t ws_size,
                              hipStream_t stream) {
    (void)n_in; (void)out_size; (void)d_ws; (void)ws_size;
    const _Float16* x   = (const _Float16*)d_in[0];
    const int*      wq  = (const int*)d_in[1];
    const _Float16* sc  = (const _Float16*)d_in[2];
    const _Float16* bs  = (const _Float16*)d_in[3];
    float*          out = (float*)d_out;

    const int Mtotal = in_sizes[0] / K_DIM;          // 4*2048 = 8192
    dim3 grid(N_DIM / BN, Mtotal / BM);              // (86, 32)
    hipLaunchKernelGGL(q4gemm_wmma_f16, grid, dim3(256), 0, stream,
                       x, wq, sc, bs, out);
}
